// SequenceAttention_13932873908954
// MI455X (gfx1250) — matrix operand order repaired and checked by reading, not compiled, not run on hardware
//
#include <hip/hip_runtime.h>
#include <hip/hip_bf16.h>

#define NNODE 2048
#define DMODEL 256
#define DSEQ 1280
#define DH 48
#define DHP 64          // padded head dim
#define NH 8
#define HDH 384
#define BG 8
#define SLEN 1024

typedef __attribute__((ext_vector_type(16))) __bf16 v16bf;
typedef __attribute__((ext_vector_type(8)))  __bf16 v8bf;
typedef __attribute__((ext_vector_type(8)))  float  v8f;

__device__ __forceinline__ v8f wmma_bf16(v16bf a, v16bf b, v8f c) {
    return __builtin_amdgcn_wmma_f32_16x16x32_bf16(false, a, false, b, (short)0, c, false, false);
}

// A-fragment (16-bit, 16x32): lane holds K = (e&7)+8*hi and +16 -> two 16B runs
__device__ __forceinline__ v16bf load_afrag(const __bf16* p, int hi) {
    v8bf lo = *(const v8bf*)(p + 8 * hi);
    v8bf up = *(const v8bf*)(p + 16 + 8 * hi);
    return __builtin_shufflevector(lo, up, 0, 1, 2, 3, 4, 5, 6, 7,
                                   8, 9, 10, 11, 12, 13, 14, 15);
}
// B-fragment (32x16): lane holds K = (e&7)+8*hi and +16 -> two 16B runs
__device__ __forceinline__ v16bf load_bfrag(const __bf16* p, int hi) {
    v8bf lo = *(const v8bf*)(p + 8 * hi);
    v8bf up = *(const v8bf*)(p + 16 + 8 * hi);
    return __builtin_shufflevector(lo, up, 0, 1, 2, 3, 4, 5, 6, 7,
                                   8, 9, 10, 11, 12, 13, 14, 15);
}

// ---------------- LayerNorm: node[2048,256] -> y bf16 ----------------
__global__ __launch_bounds__(256) void ln_kernel(const float* __restrict__ node,
                                                 const float* __restrict__ g,
                                                 const float* __restrict__ b,
                                                 __bf16* __restrict__ ybf) {
    __shared__ float red[256];
    int n = blockIdx.x, t = threadIdx.x;
    float x = node[n * DMODEL + t];
    red[t] = x; __syncthreads();
    for (int s = 128; s > 0; s >>= 1) { if (t < s) red[t] += red[t + s]; __syncthreads(); }
    float mu = red[0] / (float)DMODEL;
    __syncthreads();
    float d = x - mu;
    red[t] = d * d; __syncthreads();
    for (int s = 128; s > 0; s >>= 1) { if (t < s) red[t] += red[t + s]; __syncthreads(); }
    float var = red[0] / (float)DMODEL;
    float y = d * rsqrtf(var + 1e-5f) * g[t] + b[t];
    ybf[(size_t)n * DMODEL + t] = (__bf16)y;
}

// ---------------- elementwise converters ----------------
__global__ __launch_bounds__(256) void f2bf_kernel(const float* __restrict__ s,
                                                   __bf16* __restrict__ d, int n) {
    int i = blockIdx.x * 256 + threadIdx.x;
    if (i < n) d[i] = (__bf16)s[i];
}

// W[K,N] f32 -> Wt[N,K] bf16
__global__ __launch_bounds__(256) void wtrans_kernel(const float* __restrict__ W,
                                                     __bf16* __restrict__ Wt, int K, int N) {
    int i = blockIdx.x * 256 + threadIdx.x;
    if (i < K * N) {
        int k = i / N, n = i % N;
        Wt[(size_t)n * K + k] = (__bf16)W[i];
    }
}

__global__ __launch_bounds__(256) void fgmul_kernel(const float* __restrict__ gate,
                                                    const float* __restrict__ feat,
                                                    __bf16* __restrict__ fg, int n) {
    int i = blockIdx.x * 256 + threadIdx.x;
    if (i < n) fg[i] = (__bf16)(gate[i] * feat[i]);
}

// ---------------- batch segment starts ----------------
__global__ __launch_bounds__(256) void seg_kernel(const int* __restrict__ batch,
                                                  int* __restrict__ seg, int n) {
    __shared__ int cnt[BG];
    int t = threadIdx.x;
    if (t < BG) cnt[t] = 0;
    __syncthreads();
    for (int i = t; i < n; i += 256) atomicAdd(&cnt[batch[i]], 1);
    __syncthreads();
    if (t == 0) {
        int acc = 0;
        for (int b = 0; b < BG; b++) { seg[b] = acc; acc += cnt[b]; }
        seg[BG] = acc;
    }
}

// ---------------- generic bf16 WMMA GEMM: C[M,N] = A[M,K] x Bt[N,K]^T --------
// MODE 1: float sigmoid(acc + bias[n])
// MODE 2: bf16 store transposed for V: out[b][h][d][s]  (m=b*1024+s, n=h*48+d)
// MODE 3: float acc + bias[n] + resid[m,n]
// MODE 4: bf16 store head-padded: out[m][h][64]  (n=h*48+d, d<48; pads pre-zeroed)
template <int MODE>
__global__ __launch_bounds__(256) void gemm_bf16_kernel(const __bf16* __restrict__ A,
                                                        const __bf16* __restrict__ Bt,
                                                        const float* __restrict__ bias,
                                                        const float* __restrict__ resid,
                                                        void* __restrict__ out,
                                                        int M, int N, int K) {
    int wave = threadIdx.x >> 5;
    int l = threadIdx.x & 31, lm = l & 15, hi = l >> 4;
    int m0 = blockIdx.y * 32 + (wave >> 2) * 16;
    int n0 = blockIdx.x * 64 + (wave & 3) * 16;
    const __bf16* arow = A + (size_t)(m0 + lm) * K;
    const __bf16* brow = Bt + (size_t)(n0 + lm) * K;
    v8f acc = {};
    for (int k0 = 0; k0 < K; k0 += 32) {
        v16bf af = load_afrag(arow + k0, hi);
        v16bf bf = load_bfrag(brow + k0, hi);
        acc = wmma_bf16(af, bf, acc);
    }
    int n = n0 + lm;
    #pragma unroll
    for (int r = 0; r < 8; r++) {
        int m = m0 + r + 8 * hi;
        float v = acc[r];
        if (MODE == 1) {
            ((float*)out)[(size_t)m * N + n] = 1.f / (1.f + __expf(-(v + bias[n])));
        } else if (MODE == 2) {
            int bb = m >> 10, s = m & 1023;
            int hh = n / DH, d = n % DH;
            ((__bf16*)out)[((((size_t)bb * NH + hh) * DH + d) << 10) + s] = (__bf16)v;
        } else if (MODE == 3) {
            ((float*)out)[(size_t)m * N + n] = v + bias[n] + resid[(size_t)m * N + n];
        } else {
            int hh = n / DH, d = n % DH;
            ((__bf16*)out)[((size_t)m * NH + hh) * DHP + d] = (__bf16)v;
        }
    }
}

// ---------------- flash attention: 1 block = 16 nodes, 1 wave = 1 head ------
__global__ __launch_bounds__(256) void attn_kernel(const __bf16* __restrict__ qpad, // [2048,8,64]
                                                   const __bf16* __restrict__ kpad, // [8,1024,8,64]
                                                   const __bf16* __restrict__ vt,   // [8,8,48,1024]
                                                   const int* __restrict__ mask,    // [8,1024]
                                                   const int* __restrict__ seg,     // [9]
                                                   float* __restrict__ feat) {      // [2048,384]
    __shared__ __bf16 plds[NH][16][32];
    int b = blockIdx.x;
    int start = seg[b], end = seg[b + 1];
    int node0 = start + blockIdx.y * 16;
    if (node0 >= end) return;
    int nvalid = min(16, end - node0);

    int h = threadIdx.x >> 5;                 // wave == head
    int l = threadIdx.x & 31, lm = l & 15, hi = l >> 4;

    // Q A-fragments (two K=32 fragments over padded D_HEAD=64)
    bool rowok = lm < nvalid;
    int nodec = rowok ? (node0 + lm) : node0;   // clamp OOB rows to a safe address
    const __bf16* qrow = qpad + ((size_t)nodec * NH + h) * DHP;
    v16bf a0 = load_afrag(qrow, hi);
    v16bf a1 = load_afrag(qrow + 32, hi);
    if (!rowok) {
        v16bf z = {};
        a0 = z; a1 = z;
    }

    float rmax[8], rsum[8];
    v8f o[3] = {};
    #pragma unroll
    for (int r = 0; r < 8; r++) { rmax[r] = -3.4e38f; rsum[r] = 0.f; }

    const float inv_scale = 0.1443375673f;    // 1/sqrt(48)
    const __bf16* kb = kpad + ((size_t)b * SLEN * NH + h) * DHP;
    const __bf16* vb = vt + ((size_t)b * NH + h) * DH * SLEN;
    const int* mrow = mask + b * SLEN;

    for (int s0 = 0; s0 < SLEN; s0 += 32) {
        if (s0 + 32 < SLEN) {   // prefetch next sequence tile (K rows + V runs)
            __builtin_prefetch(kb + (size_t)(s0 + 32 + lm) * NH * DHP, 0, 0);
            __builtin_prefetch(vb + (size_t)lm * SLEN + s0 + 32, 0, 0);
        }
        v8f c[2];
        float mv[2];
        #pragma unroll
        for (int j = 0; j < 2; j++) {
            int s = s0 + j * 16 + lm;
            const __bf16* krow = kb + (size_t)s * NH * DHP;
            v16bf b0 = load_bfrag(krow, hi);
            v16bf b1 = load_bfrag(krow + 32, hi);
            v8f cc = {};
            cc = wmma_bf16(a0, b0, cc);
            cc = wmma_bf16(a1, b1, cc);
            c[j] = cc;
            mv[j] = (float)mrow[s];
        }
        // online softmax per C-row (m = r + 8*hi), columns live across 16 lanes
        #pragma unroll
        for (int r = 0; r < 8; r++) {
            float sc0 = c[0][r] * inv_scale;
            float sc1 = c[1][r] * inv_scale;
            float t = fmaxf(mv[0] > 0.f ? sc0 : -3.4e38f, mv[1] > 0.f ? sc1 : -3.4e38f);
            #pragma unroll
            for (int off = 1; off < 16; off <<= 1) t = fmaxf(t, __shfl_xor(t, off, 32));
            float nmax = fmaxf(rmax[r], t);
            float resc, p0, p1;
            if (nmax < -1e30f) { resc = 1.f; p0 = 0.f; p1 = 0.f; }
            else {
                resc = (rmax[r] < -1e30f) ? 0.f : __expf(rmax[r] - nmax);
                p0 = mv[0] > 0.f ? __expf(sc0 - nmax) : 0.f;
                p1 = mv[1] > 0.f ? __expf(sc1 - nmax) : 0.f;
                rmax[r] = nmax;
            }
            float ts = p0 + p1;
            #pragma unroll
            for (int off = 1; off < 16; off <<= 1) ts += __shfl_xor(ts, off, 32);
            rsum[r] = rsum[r] * resc + ts;
            o[0][r] *= resc; o[1][r] *= resc; o[2][r] *= resc;
            int m = r + 8 * hi;
            plds[h][m][lm] = (__bf16)p0;
            plds[h][m][16 + lm] = (__bf16)p1;
        }
        asm volatile("s_wait_dscnt 0" ::: "memory");
        // P A-fragment (16 nodes x 32 seq) straight from per-wave LDS tile
        v16bf ap = load_afrag(&plds[h][lm][0], hi);
        // accumulate O += P @ V  (3 column blocks of 16 covering D_HEAD=48)
        #pragma unroll
        for (int nb = 0; nb < 3; nb++) {
            const __bf16* vrow = vb + (size_t)(nb * 16 + lm) * SLEN + s0;
            o[nb] = wmma_bf16(ap, load_bfrag(vrow, hi), o[nb]);
        }
    }

    #pragma unroll
    for (int r = 0; r < 8; r++) {
        int m = r + 8 * hi;
        if (m < nvalid) {
            float inv = 1.f / (rsum[r] + 1e-9f);
            float* frow = feat + (size_t)(node0 + m) * HDH + h * DH;
            frow[0 * 16 + lm] = o[0][r] * inv;
            frow[1 * 16 + lm] = o[1][r] * inv;
            frow[2 * 16 + lm] = o[2][r] * inv;
        }
    }
}

// ---------------- host side ----------------
static inline size_t align256(size_t x) { return (x + 255) & ~(size_t)255; }

extern "C" void kernel_launch(void* const* d_in, const int* in_sizes, int n_in,
                              void* d_out, int out_size, void* d_ws, size_t ws_size,
                              hipStream_t stream) {
    const float* node  = (const float*)d_in[0];
    const float* emb   = (const float*)d_in[1];
    const int*   mask  = (const int*)d_in[2];
    const int*   batch = (const int*)d_in[3];
    const float* ln_g  = (const float*)d_in[4];
    const float* ln_b  = (const float*)d_in[5];
    const float* Wq    = (const float*)d_in[6];
    const float* Wk    = (const float*)d_in[7];
    const float* Wv    = (const float*)d_in[8];
    const float* Wg    = (const float*)d_in[9];
    const float* bg    = (const float*)d_in[10];
    const float* Wback = (const float*)d_in[11];
    const float* bback = (const float*)d_in[12];
    float* out = (float*)d_out;

    char* ws = (char*)d_ws;
    size_t off = 0;
    auto alloc = [&](size_t bytes) { char* p = ws + off; off += align256(bytes); return p; };

    __bf16* ybf   = (__bf16*)alloc((size_t)NNODE * DMODEL * 2);
    __bf16* embbf = (__bf16*)alloc((size_t)BG * SLEN * DSEQ * 2);
    __bf16* wq_t  = (__bf16*)alloc((size_t)HDH * DMODEL * 2);
    __bf16* wk_t  = (__bf16*)alloc((size_t)HDH * DSEQ * 2);
    __bf16* wv_t  = (__bf16*)alloc((size_t)HDH * DSEQ * 2);
    __bf16* wg_t  = (__bf16*)alloc((size_t)HDH * DMODEL * 2);
    __bf16* wb_t  = (__bf16*)alloc((size_t)DMODEL * HDH * 2);
    size_t qpad_bytes = (size_t)NNODE * NH * DHP * 2;
    size_t kpad_bytes = (size_t)BG * SLEN * NH * DHP * 2;
    __bf16* qpad  = (__bf16*)alloc(qpad_bytes);
    __bf16* kpad  = (__bf16*)alloc(kpad_bytes);
    __bf16* vtbf  = (__bf16*)alloc((size_t)BG * NH * DH * SLEN * 2);
    float*  gate  = (float*)alloc((size_t)NNODE * HDH * 4);
    float*  feat  = (float*)alloc((size_t)NNODE * HDH * 4);
    __bf16* fgbf  = (__bf16*)alloc((size_t)NNODE * HDH * 2);
    int*    seg   = (int*)alloc(16 * 4);

    // zero pad regions of head-padded Q/K (graph-capture-safe stream op)
    hipMemsetAsync(qpad, 0, qpad_bytes, stream);
    hipMemsetAsync(kpad, 0, kpad_bytes, stream);

    // 1. layernorm -> y bf16
    ln_kernel<<<NNODE, 256, 0, stream>>>(node, ln_g, ln_b, ybf);
    // 2. emb -> bf16
    {
        int n = BG * SLEN * DSEQ;
        f2bf_kernel<<<(n + 255) / 256, 256, 0, stream>>>(emb, embbf, n);
    }
    // 3. transposed bf16 weights
    wtrans_kernel<<<(DMODEL * HDH + 255) / 256, 256, 0, stream>>>(Wq, wq_t, DMODEL, HDH);
    wtrans_kernel<<<(DSEQ * HDH + 255) / 256, 256, 0, stream>>>(Wk, wk_t, DSEQ, HDH);
    wtrans_kernel<<<(DSEQ * HDH + 255) / 256, 256, 0, stream>>>(Wv, wv_t, DSEQ, HDH);
    wtrans_kernel<<<(DMODEL * HDH + 255) / 256, 256, 0, stream>>>(Wg, wg_t, DMODEL, HDH);
    wtrans_kernel<<<(HDH * DMODEL + 255) / 256, 256, 0, stream>>>(Wback, wb_t, HDH, DMODEL);
    // 4. batch segments
    seg_kernel<<<1, 256, 0, stream>>>(batch, seg, NNODE);
    // 5. projections (all WMMA)
    gemm_bf16_kernel<4><<<dim3(HDH / 64, NNODE / 32), 256, 0, stream>>>(ybf, wq_t, nullptr, nullptr, qpad, NNODE, HDH, DMODEL);
    gemm_bf16_kernel<1><<<dim3(HDH / 64, NNODE / 32), 256, 0, stream>>>(ybf, wg_t, bg, nullptr, gate, NNODE, HDH, DMODEL);
    gemm_bf16_kernel<4><<<dim3(HDH / 64, (BG * SLEN) / 32), 256, 0, stream>>>(embbf, wk_t, nullptr, nullptr, kpad, BG * SLEN, HDH, DSEQ);
    gemm_bf16_kernel<2><<<dim3(HDH / 64, (BG * SLEN) / 32), 256, 0, stream>>>(embbf, wv_t, nullptr, nullptr, vtbf, BG * SLEN, HDH, DSEQ);
    // 6. flash attention
    attn_kernel<<<dim3(BG, NNODE / 16), 256, 0, stream>>>(qpad, kpad, vtbf, mask, seg, feat);
    // 7. gate * feat -> bf16
    {
        int n = NNODE * HDH;
        fgmul_kernel<<<(n + 255) / 256, 256, 0, stream>>>(gate, feat, fgbf, n);
    }
    // 8. back projection + bias + residual
    gemm_bf16_kernel<3><<<dim3(DMODEL / 64, NNODE / 32), 256, 0, stream>>>(fgbf, wb_t, bback, node, out, NNODE, DMODEL, HDH);
    (void)in_sizes; (void)n_in; (void)out_size; (void)ws_size;
}
